// CapeGCN_46119358824987
// MI455X (gfx1250) — matrix operand order repaired and checked by reading, not compiled, not run on hardware
//
#include <hip/hip_runtime.h>
#include <hip/hip_bf16.h>

#define D 128
#define SOFTC 0.2f
#define THETAC 0.01f
#define WKC 10.0f
#define NEGK 100
#define MAXROW 10240

typedef __attribute__((ext_vector_type(16))) __bf16 bv16;
typedef __attribute__((ext_vector_type(8)))  float v8f;

typedef __attribute__((address_space(3))) unsigned short lds_us_t;

__device__ __forceinline__ unsigned short f2bf(float f) {
  unsigned int u = __float_as_uint(f);
  u += 0x7FFFu + ((u >> 16) & 1u);          // round-to-nearest-even
  return (unsigned short)(u >> 16);
}

__device__ __forceinline__ unsigned lds_off(const unsigned short* p) {
  return (unsigned)(unsigned long long)(lds_us_t*)p;   // addrspace(3) ptr -> LDS byte offset
}

// CDNA5 async DMA: 16B global -> LDS, tracked by ASYNCcnt
__device__ __forceinline__ void async_cp16(unsigned lds, const void* g) {
  asm volatile("global_load_async_to_lds_b128 %0, %1, off" :: "v"(lds), "v"(g) : "memory");
}
__device__ __forceinline__ void wait_async0() {
  asm volatile("s_wait_asynccnt 0x0" ::: "memory");
}

// ---------------- elementwise / utility kernels ----------------

__global__ void k_zero(float* p, long n) {
  long i = (long)blockIdx.x * 256 + threadIdx.x;
  if (i < n) p[i] = 0.f;
}

__global__ void k_cvt(const float* __restrict__ x, unsigned short* __restrict__ y, long n) {
  long i = (long)blockIdx.x * 256 + threadIdx.x;
  if (i < n) y[i] = f2bf(x[i]);
}

// transpose f32: in[R,C] -> out[C,R]
__global__ void k_transpose(const float* __restrict__ in, float* __restrict__ out, int R, int C) {
  __shared__ float tile[32][33];
  int c0 = blockIdx.x * 32, r0 = blockIdx.y * 32;
  int tx = threadIdx.x & 31, ty = threadIdx.x >> 5;   // blockDim 256
  for (int rr = ty; rr < 32; rr += 8) {
    int r = r0 + rr, c = c0 + tx;
    tile[rr][tx] = (r < R && c < C) ? in[(long)r * C + c] : 0.f;
  }
  __syncthreads();
  for (int rr = ty; rr < 32; rr += 8) {
    int oc = c0 + rr, orow = r0 + tx;
    if (oc < C && orow < R) out[(long)oc * R + orow] = tile[tx][rr];
  }
}

// transpose + convert to bf16: in[R,C] f32 -> out[C,R] bf16 (small weights only)
__global__ void k_transcvt(const float* __restrict__ in, unsigned short* __restrict__ out, int R, int C) {
  long i = (long)blockIdx.x * 256 + threadIdx.x;
  if (i < (long)R * C) {
    int r = (int)(i / C), c = (int)(i % C);
    out[(long)c * R + r] = f2bf(in[i]);
  }
}

// out[i*128+j] = sum_n X[n,i]*Y[n,j]   (DxD TN reduction, K = N)
__global__ void k_tn(const float* __restrict__ X, const float* __restrict__ Y,
                     float* __restrict__ out, int N) {
  int j = blockIdx.x * 16 + (threadIdx.x & 15);
  int i = blockIdx.y * 16 + (threadIdx.x >> 4);
  float s = 0.f;
  for (int n = 0; n < N; ++n) s = fmaf(X[(long)n * D + i], Y[(long)n * D + j], s);
  out[(long)i * D + j] = s;
}

// sparse segment-sum scatter: out[row[e],:] += val[e]*x[col[e],:]
__global__ void k_scatter(const int* __restrict__ row, const int* __restrict__ col,
                          const float* __restrict__ val, const float* __restrict__ x,
                          float* __restrict__ out) {
  int e = blockIdx.x;
  int r = row[e], c = col[e];
  float v = val[e];
  int d = threadIdx.x;
  atomicAdd(&out[(long)r * D + d], v * x[(long)c * D + d]);
}

// y = scale * x / max(||x||,1e-12) per row  (blockDim 128)
__global__ void k_l2n(const float* __restrict__ x, float* __restrict__ y, float scale) {
  int r = blockIdx.x, d = threadIdx.x;
  __shared__ float sh[128];
  float v = x[(long)r * D + d];
  sh[d] = v * v; __syncthreads();
  for (int s = 64; s > 0; s >>= 1) { if (d < s) sh[d] += sh[d + s]; __syncthreads(); }
  float nrm = fmaxf(sqrtf(sh[0]), 1e-12f);
  y[(long)r * D + d] = scale * v / nrm;
}

// y = base + SOFT * x/max(||x||,eps) per row
__global__ void k_recombine(const float* __restrict__ base, const float* __restrict__ x,
                            float* __restrict__ y) {
  int r = blockIdx.x, d = threadIdx.x;
  __shared__ float sh[128];
  float v = x[(long)r * D + d];
  sh[d] = v * v; __syncthreads();
  for (int s = 64; s > 0; s >>= 1) { if (d < s) sh[d] += sh[d + s]; __syncthreads(); }
  float nrm = fmaxf(sqrtf(sh[0]), 1e-12f);
  y[(long)r * D + d] = base[(long)r * D + d] + SOFTC * v / nrm;
}

__global__ void k_mean3(const float* a, const float* b, const float* c, float* y, long n) {
  long i = (long)blockIdx.x * 256 + threadIdx.x;
  if (i < n) y[i] = (a[i] + b[i] + c[i]) * (1.f / 3.f);
}

// bf16 concat [ie|im|tx] -> [N,3D]
__global__ void k_cat3(const float* a, const float* b, const float* c, unsigned short* y) {
  int i = blockIdx.x, d = threadIdx.x;
  y[(long)i * (3 * D) + d]          = f2bf(a[(long)i * D + d]);
  y[(long)i * (3 * D) + D + d]      = f2bf(b[(long)i * D + d]);
  y[(long)i * (3 * D) + 2 * D + d]  = f2bf(c[(long)i * D + d]);
}

// seq[r,:] = idx==0 ? 0 : mix[idx-1,:]
__global__ void k_gather(const int* __restrict__ idx, const float* __restrict__ mix,
                         float* __restrict__ seq) {
  int r = blockIdx.x, d = threadIdx.x;
  int k = idx[r];
  seq[(long)r * D + d] = (k == 0) ? 0.f : mix[(long)(k - 1) * D + d];
}

__global__ void k_hs(const float* __restrict__ seq, const float* __restrict__ slen,
                     float* __restrict__ hs, int L) {
  int b = blockIdx.x, d = threadIdx.x;
  float s = 0.f;
  for (int l = 0; l < L; ++l) s += seq[((long)b * L + l) * D + d];
  hs[(long)b * D + d] = s / slen[b];
}

// cat[r] = [pos_emb[l], seq[r]] in bf16
__global__ void k_catpos(const float* __restrict__ pos, const float* __restrict__ seq,
                         unsigned short* __restrict__ cat, int L) {
  int r = blockIdx.x, d = threadIdx.x;
  int l = r % L;
  cat[(long)r * (2 * D) + d]     = f2bf(pos[(long)l * D + d]);
  cat[(long)r * (2 * D) + D + d] = f2bf(seq[(long)r * D + d]);
}

// beta = (nh @ w2)*mask ; select[b] = sum_l beta*seq
__global__ void k_betasel(const float* __restrict__ nh, const float* __restrict__ seq,
                          const float* __restrict__ w2, const float* __restrict__ mask,
                          float* __restrict__ sel, int L) {
  int b = blockIdx.x, d = threadIdx.x;
  __shared__ float sh[128];
  float acc = 0.f;
  for (int l = 0; l < L; ++l) {
    long r = (long)b * L + l;
    sh[d] = nh[r * D + d] * w2[d];
    __syncthreads();
    for (int s = 64; s > 0; s >>= 1) { if (d < s) sh[d] += sh[d + s]; __syncthreads(); }
    float beta = sh[0] * mask[r];
    acc += beta * seq[r * D + d];
    __syncthreads();
  }
  sel[(long)b * D + d] = acc;
}

// fused _con_half row kernel: logits in LDS, softmax stats, top-100 by repeated argmax
__global__ __launch_bounds__(256) void k_con(const float* __restrict__ a,
                                             const float* __restrict__ raT,
                                             int N, float* __restrict__ acc) {
  __shared__ float logits[MAXROW];
  __shared__ float avec[D];
  __shared__ float red[256];
  __shared__ int redi[256];
  __shared__ float topsum_sh;
  int i = blockIdx.x, t = threadIdx.x;
  int NN = (N > MAXROW) ? MAXROW : N;
  if (t < D) avec[t] = a[(long)i * D + t];
  __syncthreads();
  for (int j = t; j < NN; j += 256) {
    float s = 0.f;
    #pragma unroll 8
    for (int d0 = 0; d0 < D; ++d0) s = fmaf(avec[d0], raT[(long)d0 * N + j], s);
    logits[j] = s;
  }
  __syncthreads();
  float m = -1e30f;
  for (int j = t; j < NN; j += 256) m = fmaxf(m, logits[j]);
  red[t] = m; __syncthreads();
  for (int s = 128; s > 0; s >>= 1) { if (t < s) red[t] = fmaxf(red[t], red[t + s]); __syncthreads(); }
  m = red[0]; __syncthreads();
  float se = 0.f;
  for (int j = t; j < NN; j += 256) se += __expf(logits[j] - m);
  red[t] = se; __syncthreads();
  for (int s = 128; s > 0; s >>= 1) { if (t < s) red[t] += red[t + s]; __syncthreads(); }
  float S = red[0]; __syncthreads();
  float diag = __expf(__expf(logits[i] - m) / S);   // sim_ii = exp(softmax_ii)
  if (t == 0) topsum_sh = 0.f;
  __syncthreads();
  for (int it = 0; it < NEGK; ++it) {
    float lm = -1e30f; int li = 0;
    for (int j = t; j < NN; j += 256) { float v = logits[j]; if (v > lm) { lm = v; li = j; } }
    red[t] = lm; redi[t] = li; __syncthreads();
    for (int s = 128; s > 0; s >>= 1) {
      if (t < s && red[t + s] > red[t]) { red[t] = red[t + s]; redi[t] = redi[t + s]; }
      __syncthreads();
    }
    if (t == 0) { topsum_sh += __expf(__expf(red[0] - m) / S); logits[redi[0]] = -1e30f; }
    __syncthreads();
  }
  if (t == 0) atomicAdd(acc, log10f(topsum_sh) - log10f(diag));
}

__global__ void k_final(const float* __restrict__ conacc, float* __restrict__ out, long off) {
  out[off] = THETAC * conacc[0];
}

// ---------------- WMMA bf16 GEMM (NT): C[M,Nc] = A[M,K] @ B[Nc,K]^T ----------------
// Tiles staged with GLOBAL_LOAD_ASYNC_TO_LDS_B128 (ASYNCcnt), double-buffered LDS.
// act: 0 none, 1 tanh, 2 sigmoid. optional bias[col], optional extra[(row/extraDiv),col].
#define LDT 40   // row stride in ushorts (80B: 16B aligned, non-power-of-2 for banking)
__global__ __launch_bounds__(128)
void k_gemm(const unsigned short* __restrict__ A, const unsigned short* __restrict__ Bm,
            const float* __restrict__ bias, const float* __restrict__ extra, int extraDiv,
            float* __restrict__ C, int M, int Nc, int K, int act) {
  __shared__ unsigned short As[2][32 * LDT];
  __shared__ unsigned short Bs[2][64 * LDT];
  const int tid = threadIdx.x;
  const int lane = tid & 31;
  const int wave = tid >> 5;
  const int m0 = blockIdx.y * 32;
  const int n0 = blockIdx.x * 64;
  const int wm = (wave & 1) * 16;
  const int wn = (wave >> 1) * 32;
  const int hlf = lane >> 4;
  const int l15 = lane & 15;

  // per-thread async-copy assignments: A tile 32x32 ushort = 128 x 16B chunks,
  // B tile 64x32 ushort = 256 x 16B chunks (2 per thread)
  const int ra  = tid >> 2;            // 0..31
  const int ca  = (tid & 3) * 8;       // ushort offset of 16B chunk in k-slice
  const int rb0 = tid >> 2;            // 0..31
  const int rb1 = (tid + 128) >> 2;    // 32..63
  const bool aok  = (m0 + ra)  < M;
  const bool b0ok = (n0 + rb0) < Nc;
  const bool b1ok = (n0 + rb1) < Nc;
  const unsigned short* gA  = A  + (long)(m0 + ra)  * K + ca;
  const unsigned short* gB0 = Bm + (long)(n0 + rb0) * K + ca;
  const unsigned short* gB1 = Bm + (long)(n0 + rb1) * K + ca;
  unsigned lA[2], lB0[2], lB1[2];
  #pragma unroll
  for (int b = 0; b < 2; ++b) {
    lA[b]  = lds_off(&As[b][ra  * LDT + ca]);
    lB0[b] = lds_off(&Bs[b][rb0 * LDT + ca]);
    lB1[b] = lds_off(&Bs[b][rb1 * LDT + ca]);
  }
  // OOB lanes simply skip the copy: stale LDS only feeds C rows/cols that the
  // guarded epilogue never stores, so no zero-fill is needed and EXEC is all-1
  // at every WMMA.
  auto issue = [&](int k0, int buf) {
    if (aok)  async_cp16(lA[buf],  gA  + k0);
    if (b0ok) async_cp16(lB0[buf], gB0 + k0);
    if (b1ok) async_cp16(lB1[buf], gB1 + k0);
  };

  v8f acc0 = {0.f, 0.f, 0.f, 0.f, 0.f, 0.f, 0.f, 0.f};
  v8f acc1 = {0.f, 0.f, 0.f, 0.f, 0.f, 0.f, 0.f, 0.f};

  int cur = 0;
  issue(0, 0);
  for (int k0 = 0; k0 < K; k0 += 32) {
    wait_async0();          // my tile-copies done
    __syncthreads();        // everyone's tile-copies done
    if (k0 + 32 < K) issue(k0 + 32, cur ^ 1);   // prefetch next tile into other buffer

    const unsigned short* as = &As[cur][0];
    const unsigned short* bs = &Bs[cur][0];
    union { unsigned int u[8]; bv16 v; } fa, fb0, fb1;
    #pragma unroll
    for (int p = 0; p < 8; ++p) {
      // A frag: lane<16 holds K {0..7,16..23}; lane>=16 holds K {8..15,24..31}
      int ka = (p < 4) ? (hlf * 8 + 2 * p) : (hlf * 8 + 16 + 2 * (p - 4));
      // B frag: same K order as A frag
      int kb = ka;
      fa.u[p]  = *(const unsigned int*)(as + (wm + l15) * LDT + ka);
      fb0.u[p] = *(const unsigned int*)(bs + (wn + l15) * LDT + kb);
      fb1.u[p] = *(const unsigned int*)(bs + (wn + 16 + l15) * LDT + kb);
    }
    acc0 = __builtin_amdgcn_wmma_f32_16x16x32_bf16(false, fa.v, false, fb0.v, (short)0, acc0, false, false);
    acc1 = __builtin_amdgcn_wmma_f32_16x16x32_bf16(false, fa.v, false, fb1.v, (short)0, acc1, false, false);
    cur ^= 1;
  }

  const int rb = (lane < 16) ? 0 : 8;
  #pragma unroll
  for (int r = 0; r < 8; ++r) {
    int row = m0 + wm + rb + r;
    if (row >= M) continue;
    long rowOff = (long)row * Nc;
    long eoff = extra ? ((long)(row / extraDiv) * Nc) : 0;
    for (int g = 0; g < 2; ++g) {
      int col = n0 + wn + g * 16 + l15;
      if (col >= Nc) continue;
      float x = (g == 0) ? acc0[r] : acc1[r];
      if (bias)  x += bias[col];
      if (extra) x += extra[eoff + col];
      if (act == 1) x = tanhf(x);
      else if (act == 2) x = 1.f / (1.f + __expf(-x));
      C[rowOff + col] = x;
    }
  }
}

// ---------------- host orchestration ----------------

extern "C" void kernel_launch(void* const* d_in, const int* in_sizes, int n_in,
                              void* d_out, int out_size, void* d_ws, size_t ws_size,
                              hipStream_t stream) {
  (void)n_in; (void)out_size; (void)ws_size;
  const float* adj_val = (const float*)d_in[0];
  const float* emb     = (const float*)d_in[1];
  const float* img     = (const float*)d_in[2];
  const float* txt     = (const float*)d_in[3];
  const float* pos     = (const float*)d_in[4];
  const float* w_item  = (const float*)d_in[5];
  const float* w1      = (const float*)d_in[6];
  const float* w2      = (const float*)d_in[7];
  const float* glu1_w  = (const float*)d_in[8];
  const float* glu1_b  = (const float*)d_in[9];
  const float* glu2_w  = (const float*)d_in[10];
  const float* pimg_w  = (const float*)d_in[11];
  const float* pimg_b  = (const float*)d_in[12];
  const float* ptxt_w  = (const float*)d_in[13];
  const float* ptxt_b  = (const float*)d_in[14];
  const float* m1_w    = (const float*)d_in[15];
  const float* m1_b    = (const float*)d_in[16];
  const float* m2_w    = (const float*)d_in[17];
  const float* m2_b    = (const float*)d_in[18];
  const float* slen    = (const float*)d_in[19];
  const float* maskp   = (const float*)d_in[20];
  const int* adj_row   = (const int*)d_in[21];
  const int* adj_col   = (const int*)d_in[22];
  const int* rsi       = (const int*)d_in[23];

  const int Nn  = in_sizes[1] / D;
  const int NNZ = in_sizes[0];
  const int Bb  = in_sizes[19];
  const int Ll  = in_sizes[20] / Bb;
  const long NB  = (long)Nn * D;
  const long BL  = (long)Bb * Ll;
  const long BLD = BL * D;

  char* wp = (char*)d_ws;
  size_t off = 0;
  auto alloc = [&](size_t bytes) -> void* {
    void* p = wp + off;
    off += (bytes + 255) & ~(size_t)255;
    return p;
  };

  float* f_pi    = (float*)alloc(NB * 4);
  float* f_pt    = (float*)alloc(NB * 4);
  float* f_tmp   = (float*)alloc(NB * 4);
  float* f_reimg = (float*)alloc(NB * 4);
  float* f_retxt = (float*)alloc(NB * 4);
  float* f_raT   = (float*)alloc(NB * 4);
  float* f_e     = (float*)alloc(NB * 4);
  float* f_n1    = (float*)alloc(NB * 4);
  float* f_n2    = (float*)alloc(NB * 4);
  float* f_ie    = (float*)alloc(NB * 4);
  float* f_im    = (float*)alloc(NB * 4);
  float* f_tx    = (float*)alloc(NB * 4);
  float* f_mix2  = (float*)alloc(NB * 4);
  float* f_Mt    = (float*)alloc(D * D * 4);
  float* f_hs    = (float*)alloc((long)Bb * D * 4);
  float* f_hsg   = (float*)alloc((long)Bb * D * 4);
  float* f_sel   = (float*)alloc((long)Bb * D * 4);
  float* f_sess  = (float*)alloc((long)Bb * D * 4);
  float* f_seq   = (float*)alloc(BLD * 4);
  float* f_nh1   = (float*)alloc(BLD * 4);
  float* f_nh2   = (float*)alloc(BLD * 4);
  float* f_con   = (float*)alloc(256);
  unsigned short* b_emb   = (unsigned short*)alloc(NB * 2);
  unsigned short* b_img   = (unsigned short*)alloc(NB * 2);
  unsigned short* b_txt   = (unsigned short*)alloc(NB * 2);
  unsigned short* b_gen   = (unsigned short*)alloc(3 * NB * 2);
  unsigned short* b_w     = (unsigned short*)alloc(3 * D * D * 2);
  unsigned short* b_Mt    = (unsigned short*)alloc(D * D * 2);
  unsigned short* b_cat   = (unsigned short*)alloc(BL * 2 * D * 2);
  unsigned short* b_small = (unsigned short*)alloc((long)Bb * D * 2);

  auto cvt  = [&](const float* x, unsigned short* y, long n) {
    k_cvt<<<(unsigned)((n + 255) / 256), 256, 0, stream>>>(x, y, n);
  };
  auto zero = [&](float* p, long n) {
    k_zero<<<(unsigned)((n + 255) / 256), 256, 0, stream>>>(p, n);
  };
  auto gemm = [&](const unsigned short* A, const unsigned short* Bm, const float* bias,
                  const float* extra, int extraDiv, float* C, int M, int Nc, int K, int act) {
    dim3 g((Nc + 63) / 64, (M + 31) / 32);
    k_gemm<<<g, 128, 0, stream>>>(A, Bm, bias, extra, extraDiv, C, M, Nc, K, act);
  };

  // bf16 activations
  cvt(emb, b_emb, NB); cvt(img, b_img, NB); cvt(txt, b_txt, NB);

  // prompt projections: p_i, p_t
  cvt(pimg_w, b_w, D * D);
  gemm(b_emb, b_w, pimg_b, nullptr, 1, f_pi, Nn, D, D, 0);
  cvt(ptxt_w, b_w, D * D);
  gemm(b_emb, b_w, ptxt_b, nullptr, 1, f_pt, Nn, D, D, 0);

  // re_img = img + SOFT*l2n(p_i @ (p_i.T @ img))  (Mt holds transposed DxD so NT-GEMM applies)
  k_tn<<<dim3(D / 16, D / 16), 256, 0, stream>>>(img, f_pi, f_Mt, Nn);
  cvt(f_Mt, b_Mt, D * D);
  cvt(f_pi, b_gen, NB);
  gemm(b_gen, b_Mt, nullptr, nullptr, 1, f_tmp, Nn, D, D, 0);
  k_recombine<<<Nn, 128, 0, stream>>>(img, f_tmp, f_reimg);
  // re_txt
  k_tn<<<dim3(D / 16, D / 16), 256, 0, stream>>>(txt, f_pt, f_Mt, Nn);
  cvt(f_Mt, b_Mt, D * D);
  cvt(f_pt, b_gen, NB);
  gemm(b_gen, b_Mt, nullptr, nullptr, 1, f_tmp, Nn, D, D, 0);
  k_recombine<<<Nn, 128, 0, stream>>>(txt, f_tmp, f_retxt);

  // contrastive term (fused per-row: softmax stats + top-100 + diagonal)
  zero(f_con, 1);
  k_transpose<<<dim3((D + 31) / 32, (Nn + 31) / 32), 256, 0, stream>>>(f_reimg, f_raT, Nn, D);
  k_con<<<Nn, 256, 0, stream>>>(img, f_raT, Nn, f_con);
  k_transpose<<<dim3((D + 31) / 32, (Nn + 31) / 32), 256, 0, stream>>>(f_retxt, f_raT, Nn, D);
  k_con<<<Nn, 256, 0, stream>>>(txt, f_raT, Nn, f_con);

  // GCN item conv per modality
  auto conv = [&](const float* src, const unsigned short* srcb, float* outp) {
    cvt(w_item, b_w, D * D);
    gemm(srcb, b_w, nullptr, nullptr, 1, f_tmp, Nn, D, D, 0);
    zero(f_e, NB);
    k_scatter<<<NNZ, 128, 0, stream>>>(adj_row, adj_col, adj_val, f_tmp, f_e);
    k_l2n<<<Nn, 128, 0, stream>>>(f_e, f_n1, 1.0f);
    cvt(f_e, b_gen, NB);
    cvt(w_item + (long)D * D, b_w, D * D);
    gemm(b_gen, b_w, nullptr, nullptr, 1, f_tmp, Nn, D, D, 0);
    zero(f_e, NB);
    k_scatter<<<NNZ, 128, 0, stream>>>(adj_row, adj_col, adj_val, f_tmp, f_e);
    k_l2n<<<Nn, 128, 0, stream>>>(f_e, f_n2, 1.0f);
    k_mean3<<<(unsigned)((NB + 255) / 256), 256, 0, stream>>>(src, f_n1, f_n2, outp, NB);
  };
  conv(emb, b_emb, f_ie);
  conv(img, b_img, f_im);
  conv(txt, b_txt, f_tx);

  // modality mixer
  k_cat3<<<Nn, 128, 0, stream>>>(f_ie, f_im, f_tx, b_gen);
  cvt(m1_w, b_w, (long)D * 3 * D);
  gemm(b_gen, b_w, m1_b, nullptr, 1, f_tmp, Nn, D, 3 * D, 1);
  cvt(f_tmp, b_gen, NB);
  cvt(m2_w, b_w, D * D);
  gemm(b_gen, b_w, m2_b, nullptr, 1, f_mix2, Nn, D, D, 1);

  // session attention
  k_gather<<<(unsigned)BL, 128, 0, stream>>>(rsi, f_mix2, f_seq);
  k_hs<<<Bb, 128, 0, stream>>>(f_seq, slen, f_hs, Ll);
  cvt(f_hs, b_small, (long)Bb * D);
  cvt(glu2_w, b_w, D * D);
  gemm(b_small, b_w, nullptr, nullptr, 1, f_hsg, Bb, D, D, 0);
  k_catpos<<<(unsigned)BL, 128, 0, stream>>>(pos, f_seq, b_cat, Ll);
  k_transcvt<<<(unsigned)((2 * D * D + 255) / 256), 256, 0, stream>>>(w1, b_w, 2 * D, D);
  gemm(b_cat, b_w, nullptr, nullptr, 1, f_nh1, (int)BL, D, 2 * D, 1);
  cvt(f_nh1, b_cat, BLD);                        // reuse b_cat as bf16 nh1
  cvt(glu1_w, b_w, D * D);
  gemm(b_cat, b_w, glu1_b, f_hsg, Ll, f_nh2, (int)BL, D, D, 2);
  k_betasel<<<Bb, 128, 0, stream>>>(f_nh2, f_seq, w2, maskp, f_sel, Ll);
  k_l2n<<<Bb, 128, 0, stream>>>(f_sel, f_sess, WKC);

  // scores = sess @ l2n(ie).T -> d_out[0 .. B*N)
  k_l2n<<<Nn, 128, 0, stream>>>(f_ie, f_tmp, 1.0f);
  cvt(f_tmp, b_emb, NB);                         // reuse b_emb as bf16 ie_norm
  cvt(f_sess, b_small, (long)Bb * D);
  gemm(b_small, b_emb, nullptr, nullptr, 1, (float*)d_out, Bb, Nn, D, 0);

  // d_out[B*N] = THETA * con
  k_final<<<1, 1, 0, stream>>>(f_con, (float*)d_out, (long)Bb * Nn);
}
